// MyLLMMoERouter_78718160601089
// MI455X (gfx1250) — matrix operand order repaired and checked by reading, not compiled, not run on hardware
//
#include <hip/hip_runtime.h>

// ---------------------------------------------------------------------------
// MoE router for MI455X (gfx1250, wave32, WMMA).
// gate = x @ W (bf16x2 split-precision on v_wmma_f32_16x16x32_bf16),
// then per-token top-8 + masked softmax.
// B=4, S=4096, H=4096, E=64, K(top)=8, TEMP=1.0
// ---------------------------------------------------------------------------

typedef __attribute__((ext_vector_type(16))) __bf16 v16bf;
typedef __attribute__((ext_vector_type(8)))  float  v8f;
typedef __attribute__((ext_vector_type(4)))  float  v4f;

#define HDIM    4096
#define EDIM    64
#define TOPK    8
#define TOKENS  16384           // B*S
#define KCHUNKS 128             // HDIM / 32
#define NTILES  4               // EDIM / 16
#define FRAG_ELEMS 16           // bf16 elems per lane per fragment
#define WPACK_ELEMS (KCHUNKS * NTILES * 32 * FRAG_ELEMS)  // 262144 bf16 per plane

// Native converts: lower to v_cvt_pk_bf16_f32 / v_fma_mix_f32_bf16 on gfx1250.
__device__ __forceinline__ __bf16 f2bf(float f)  { return (__bf16)f; }
__device__ __forceinline__ float  bf2f(__bf16 h) { return (float)h;  }

// ---------------------------------------------------------------------------
// Kernel 1: pack W [H,E] fp32 into per-lane B fragments (hi + lo bf16 planes).
// B tile = 32(K) x 16(N) bf16. Layout (dense 16-bit B, wave32):
//   lanes 0-15 : N = lane,    elems 0..7 -> K = c*32 + 0..7,  elems 8..15 -> K = c*32 + 16..23
//   lanes 16-31: N = lane-16, elems 0..7 -> K = c*32 + 8..15, elems 8..15 -> K = c*32 + 24..31
// Fragment index f = ((c*NTILES + t)*32 + lane), FRAG_ELEMS bf16 each.
// ---------------------------------------------------------------------------
__global__ __launch_bounds__(256) void moe_pack_w(const float* __restrict__ W,
                                                  __bf16* __restrict__ wHi,
                                                  __bf16* __restrict__ wLo) {
    int f = blockIdx.x * blockDim.x + threadIdx.x;
    if (f >= KCHUNKS * NTILES * 32) return;
    const int lane = f & 31;
    const int t    = (f >> 5) & (NTILES - 1);
    const int c    = f >> 7;
    const int g    = lane >> 4;
    const int e    = t * 16 + (lane & 15);
    __bf16* hp = wHi + (size_t)f * FRAG_ELEMS;
    __bf16* lp = wLo + (size_t)f * FRAG_ELEMS;
#pragma unroll
    for (int i = 0; i < FRAG_ELEMS; ++i) {
        const int k = c * 32 + (i >= 8 ? 16 : 0) + g * 8 + (i & 7);
        const float w = W[(size_t)k * EDIM + e];
        const __bf16 hi = f2bf(w);
        hp[i] = hi;
        lp[i] = f2bf(w - bf2f(hi));
    }
}

// ---------------------------------------------------------------------------
// Kernel 2: gate GEMM. One wave = 16 tokens x 64 experts (4 C tiles).
// A fragment layout (16-bit A, 16x32, wave32):
//   lanes 0-15 : M = lane,    elems 0..7  -> K = 0..7,  elems 8..15 -> K = 16..23
//   lanes 16-31: M = lane-16, elems 0..7  -> K = 8..15, elems 8..15 -> K = 24..31
// i.e. K = c*32 + (i>=8 ? 16 : 0) + g*8 + (i&7)  -> two runs of 8 contiguous K.
//
// Software-pipelined body: A loads (HBM) -> all 8 B fragment loads (L2-hot)
// -> A hi/lo bf16 converts -> 12 WMMAs round-robin over the 4 accumulators.
// ---------------------------------------------------------------------------
__global__ __launch_bounds__(128) void moe_gate_gemm(const float* __restrict__ x,
                                                     const __bf16* __restrict__ wHi,
                                                     const __bf16* __restrict__ wLo,
                                                     float* __restrict__ gate) {
    const int lane = threadIdx.x & 31;
    const int wave = threadIdx.x >> 5;
    const int g    = lane >> 4;         // lane group
    const int m    = lane & 15;
    const int tokBase = (blockIdx.x * 4 + wave) * 16;
    const size_t rowOff = (size_t)(tokBase + m) * HDIM;

    const v16bf* bhBase = (const v16bf*)wHi;
    const v16bf* blBase = (const v16bf*)wLo;

    v8f acc0 = {}, acc1 = {}, acc2 = {}, acc3 = {};

#pragma unroll 1
    for (int c = 0; c < KCHUNKS; ++c) {
        // ---- A loads first (HBM stream, longest latency) ----
        const float* p0 = x + rowOff + c * 32 + g * 8;
        v4f f0 = *(const v4f*)(p0);
        v4f f1 = *(const v4f*)(p0 + 4);
        v4f f2 = *(const v4f*)(p0 + 16);
        v4f f3 = *(const v4f*)(p0 + 20);
        __builtin_prefetch(p0 + 32, 0, 0);      // next K chunk -> global_prefetch_b8

        // ---- All 8 B fragments up front (one big clause, partial waits) ----
        const v16bf* bh = bhBase + (size_t)(c * NTILES) * 32 + lane;
        const v16bf* bl = blBase + (size_t)(c * NTILES) * 32 + lane;
        v16bf bh0 = bh[0], bh1 = bh[32], bh2 = bh[64], bh3 = bh[96];
        v16bf bl0 = bl[0], bl1 = bl[32], bl2 = bl[64], bl3 = bl[96];

        // ---- A hi/lo split to bf16 (waits only on the A loads) ----
        float va[16];
#pragma unroll
        for (int i = 0; i < 4; ++i) {
            va[i]      = f0[i];
            va[4 + i]  = f1[i];
            va[8 + i]  = f2[i];
            va[12 + i] = f3[i];
        }
        union { v16bf v; __bf16 e[16]; } ah, al;
#pragma unroll
        for (int i = 0; i < 16; ++i) {
            const __bf16 h = f2bf(va[i]);
            ah.e[i] = h;
            al.e[i] = f2bf(va[i] - bf2f(h));
        }

        // ---- 12 WMMAs, round-robin across accumulators per term ----
        acc0 = __builtin_amdgcn_wmma_f32_16x16x32_bf16(false, ah.v, false, bh0, (short)0, acc0, false, false);
        acc1 = __builtin_amdgcn_wmma_f32_16x16x32_bf16(false, ah.v, false, bh1, (short)0, acc1, false, false);
        acc2 = __builtin_amdgcn_wmma_f32_16x16x32_bf16(false, ah.v, false, bh2, (short)0, acc2, false, false);
        acc3 = __builtin_amdgcn_wmma_f32_16x16x32_bf16(false, ah.v, false, bh3, (short)0, acc3, false, false);

        acc0 = __builtin_amdgcn_wmma_f32_16x16x32_bf16(false, al.v, false, bh0, (short)0, acc0, false, false);
        acc1 = __builtin_amdgcn_wmma_f32_16x16x32_bf16(false, al.v, false, bh1, (short)0, acc1, false, false);
        acc2 = __builtin_amdgcn_wmma_f32_16x16x32_bf16(false, al.v, false, bh2, (short)0, acc2, false, false);
        acc3 = __builtin_amdgcn_wmma_f32_16x16x32_bf16(false, al.v, false, bh3, (short)0, acc3, false, false);

        acc0 = __builtin_amdgcn_wmma_f32_16x16x32_bf16(false, ah.v, false, bl0, (short)0, acc0, false, false);
        acc1 = __builtin_amdgcn_wmma_f32_16x16x32_bf16(false, ah.v, false, bl1, (short)0, acc1, false, false);
        acc2 = __builtin_amdgcn_wmma_f32_16x16x32_bf16(false, ah.v, false, bl2, (short)0, acc2, false, false);
        acc3 = __builtin_amdgcn_wmma_f32_16x16x32_bf16(false, ah.v, false, bl3, (short)0, acc3, false, false);
    }

    // C/D layout: VGPR j, lanes 0-15 -> M=j, N=lane ; lanes 16-31 -> M=j+8, N=lane-16
    const int rowBase = tokBase + 8 * g;
#pragma unroll
    for (int j = 0; j < 8; ++j) {
        float* row = gate + (size_t)(rowBase + j) * EDIM + m;
        row[0]  = acc0[j];
        row[16] = acc1[j];
        row[32] = acc2[j];
        row[48] = acc3[j];
    }
}

// ---------------------------------------------------------------------------
// Kernel 3: per-token top-8 (biased logits, ties -> lowest index, matching
// jax.lax.top_k) + softmax over the raw selected logits, scattered to E-wide
// rows (zeros elsewhere). One thread per token.
// ---------------------------------------------------------------------------
__global__ __launch_bounds__(256) void moe_topk(const float* __restrict__ gate,
                                                const float* __restrict__ b_lin,
                                                const float* __restrict__ gate_bias,
                                                float* __restrict__ outP,
                                                int*   __restrict__ outIds) {
    const int tok = blockIdx.x * blockDim.x + threadIdx.x;
    if (tok >= TOKENS) return;
    const float invT = 1.0f;                 // 1/TEMP, TEMP = 1.0
    const float* grow = gate + (size_t)tok * EDIM;

    float biased[EDIM];
    for (int e = 0; e < EDIM; ++e) {
        const float lg = (grow[e] + b_lin[e]) * invT;
        biased[e] = lg + gate_bias[e];
    }

    int   ids8[TOPK];
    float vals[TOPK];
    unsigned long long taken = 0ull;
    for (int kk = 0; kk < TOPK; ++kk) {
        float best = -__builtin_inff();
        int   bi   = 0;
        for (int e = 0; e < EDIM; ++e) {
            const bool free_e = ((taken >> e) & 1ull) == 0ull;
            if (free_e && biased[e] > best) { best = biased[e]; bi = e; }
        }
        taken |= (1ull << bi);
        ids8[kk] = bi;
        vals[kk] = (grow[bi] + b_lin[bi]) * invT;   // raw (un-biased) gate logit
    }

    float mx = vals[0];
    for (int kk = 1; kk < TOPK; ++kk) mx = fmaxf(mx, vals[kk]);
    float s = 0.f;
    float ex[TOPK];
    for (int kk = 0; kk < TOPK; ++kk) { ex[kk] = __expf(vals[kk] - mx); s += ex[kk]; }
    const float rs = 1.0f / s;

    float* prow = outP + (size_t)tok * EDIM;
    for (int e = 0; e < EDIM; ++e) prow[e] = 0.0f;
    for (int kk = 0; kk < TOPK; ++kk) {
        prow[ids8[kk]] = ex[kk] * rs;
        outIds[(size_t)tok * TOPK + kk] = ids8[kk];
    }
}

// ---------------------------------------------------------------------------
extern "C" void kernel_launch(void* const* d_in, const int* in_sizes, int n_in,
                              void* d_out, int out_size, void* d_ws, size_t ws_size,
                              hipStream_t stream) {
    (void)in_sizes; (void)n_in; (void)out_size; (void)ws_size;
    const float* x         = (const float*)d_in[0];   // [B,S,H] fp32
    const float* W         = (const float*)d_in[1];   // [H,E]   fp32
    const float* b_lin     = (const float*)d_in[2];   // [E]
    const float* gate_bias = (const float*)d_in[3];   // [E]

    // Workspace layout: [Wh bf16 512KB][Wl bf16 512KB][gate fp32 4MB]
    __bf16* wHi = (__bf16*)d_ws;
    __bf16* wLo = wHi + WPACK_ELEMS;
    float*  gateBuf = (float*)((char*)d_ws + (size_t)2 * WPACK_ELEMS * sizeof(__bf16));

    float* outP  = (float*)d_out;                       // [B,S,E] fp32
    int*   outId = (int*)(outP + (size_t)TOKENS * EDIM); // [B,S,TOPK] int32

    moe_pack_w  <<<(KCHUNKS * NTILES * 32 + 255) / 256, 256, 0, stream>>>(W, wHi, wLo);
    moe_gate_gemm<<<TOKENS / 64, 128, 0, stream>>>(x, wHi, wLo, gateBuf);
    moe_topk    <<<(TOKENS + 255) / 256, 256, 0, stream>>>(gateBuf, b_lin, gate_bias, outP, outId);
}
